// BiMambaLayer_72816875536909
// MI455X (gfx1250) — compile-verified, matrix operand order checked by reading, not run on hardware
//
#include <hip/hip_runtime.h>
#include <hip/hip_bf16.h>
#include <math.h>

// ---------------------------------------------------------------------------
// BiMamba layer for MI455X (gfx1250, wave32).
//  - in_proj / out_proj: 24576x384x384 GEMMs on v_wmma_f32_16x16x32_bf16,
//    operands pre-converted to bf16 so the hot loop is pure b128-load + wmma
//  - selective scan: chunked 3-pass associative scan (16 chunks of 128)
// ---------------------------------------------------------------------------

typedef __attribute__((ext_vector_type(16))) __bf16 v16bf;
typedef __attribute__((ext_vector_type(8)))  __bf16 v8bf;
typedef __attribute__((ext_vector_type(8)))  float  v8f;

#define TLEN   2048
#define BFN    12           // b * F
#define DIN    192          // dim/2
#define GK     384          // DIM (also K of both GEMMs)
#define RDIM   56           // DTRANK + 2*DSTATE
#define DTRANK 24
#define NSTATE 16
#define NCHUNK 16
#define CLEN   128          // TLEN / NCHUNK
#define NROW   24576        // BFN * TLEN

// ---------------------------------------------------------------------------
// fp32 -> bf16 pointwise conversion, 8 elements / thread (b128 in, b128 out)
// ---------------------------------------------------------------------------
__global__ __launch_bounds__(256) void cvt_bf16_kernel(
    const float* __restrict__ in, __bf16* __restrict__ out, size_t n)
{
    const size_t i = ((size_t)blockIdx.x * 256 + threadIdx.x) * 8;
    if (i >= n) return;
    float4 a = *(const float4*)(in + i);
    float4 b = *(const float4*)(in + i + 4);
    v8bf o;
    o[0] = (__bf16)a.x; o[1] = (__bf16)a.y; o[2] = (__bf16)a.z; o[3] = (__bf16)a.w;
    o[4] = (__bf16)b.x; o[5] = (__bf16)b.y; o[6] = (__bf16)b.z; o[7] = (__bf16)b.w;
    *(v8bf*)(out + i) = o;
}

// ---------------------------------------------------------------------------
// WMMA bf16 fragment loader from bf16 memory: 8 contiguous bf16 at p[0..7]
// (K=kb+0..7) and p[16..23] (K=16+kb+0..7).  Two b128 loads, no conversion.
// ---------------------------------------------------------------------------
union FragU { int4 q[2]; v16bf v; };

__device__ __forceinline__ v16bf ld_frag(const __bf16* __restrict__ p) {
    FragU u;
    u.q[0] = *(const int4*)(p);        // K = kb + 0..7
    u.q[1] = *(const int4*)(p + 16);   // K = kb + 16..23
    return u.v;
}

// ---------------------------------------------------------------------------
// C[M][384] = A[M][384] * W[384][384]^T + bias   (A,W bf16; C fp32)
// Block: 8 waves; wave w -> M-tile (blockIdx.x*128 + w*16), 6 N-tiles (96 cols)
// Grid: (M/128, 384/96)
// ---------------------------------------------------------------------------
__global__ __launch_bounds__(256) void wmma_gemm384(
    const __bf16* __restrict__ A, const __bf16* __restrict__ W,
    const float* __restrict__ bias, float* __restrict__ C)
{
    const int lane = threadIdx.x & 31;
    const int wave = threadIdx.x >> 5;
    const int half = lane >> 4;      // 0/1 -> K shift of 8
    const int mr   = lane & 15;      // M (A frag) or N (B frag) within tile
    const int kb   = half * 8;

    const int rowTile = blockIdx.x * 128 + wave * 16;
    const int colBase = blockIdx.y * 96;

    v8f acc[6];
#pragma unroll
    for (int n = 0; n < 6; ++n)
#pragma unroll
        for (int r = 0; r < 8; ++r) acc[n][r] = 0.0f;

    const __bf16* arow = A + (size_t)(rowTile + mr) * GK + kb;

    for (int k0 = 0; k0 < GK; k0 += 32) {
        v16bf af = ld_frag(arow + k0);
#pragma unroll
        for (int n = 0; n < 6; ++n) {
            const __bf16* brow = W + (size_t)(colBase + n * 16 + mr) * GK + kb + k0;
            v16bf bfr = ld_frag(brow);
            acc[n] = __builtin_amdgcn_wmma_f32_16x16x32_bf16(
                false, af, false, bfr, (short)0, acc[n], false, false);
        }
    }

    // D layout: VGPR r -> M = half*8 + r ; N = lane&15
#pragma unroll
    for (int n = 0; n < 6; ++n) {
        const int col = colBase + n * 16 + mr;
        const float bv = bias[col];
#pragma unroll
        for (int r = 0; r < 8; ++r) {
            const int row = rowTile + half * 8 + r;
            C[(size_t)row * GK + col] = acc[n][r] + bv;
        }
    }
}

// ---------------------------------------------------------------------------
// Grouped conv3 (SAME, groups=192, 2 in-ch per group) + SiLU.
// xp: (bf,t,384) row-major fp32 ; xc out: (bf,t,192) fp32
// ---------------------------------------------------------------------------
__global__ __launch_bounds__(192) void conv_silu_kernel(
    const float* __restrict__ xp, const float* __restrict__ cw,
    const float* __restrict__ cb, float* __restrict__ xc)
{
    const int d  = threadIdx.x;
    const int bt = blockIdx.x;           // bf*TLEN + t
    const int t  = bt & (TLEN - 1);

    float w[6];
#pragma unroll
    for (int i = 0; i < 6; ++i) w[i] = cw[d * 6 + i];   // (d, i, k) row-major

    const float* base = xp + (size_t)bt * GK + 2 * d;
    float acc = cb[d];
#pragma unroll
    for (int k = 0; k < 3; ++k) {
        const int tt = t + k - 1;
        if (tt >= 0 && tt < TLEN) {
            const float* pr = base + (long)(k - 1) * GK;
            acc += w[k] * pr[0] + w[3 + k] * pr[1];
        }
    }
    xc[(size_t)bt * DIN + d] = acc / (1.0f + __expf(-acc));   // SiLU
}

// ---------------------------------------------------------------------------
// x_dbl[bt][r] = sum_d x_proj_w[r][d] * xc[bt][d]   (r < 56)
// ---------------------------------------------------------------------------
__global__ __launch_bounds__(64) void xdbl_kernel(
    const float* __restrict__ xc, const float* __restrict__ xpw,
    float* __restrict__ xdbl)
{
    __shared__ float row[DIN];
    const int bt = blockIdx.x;
    for (int i = threadIdx.x; i < DIN; i += 64) row[i] = xc[(size_t)bt * DIN + i];
    __syncthreads();
    const int r = threadIdx.x;
    if (r < RDIM) {
        const float* wr = xpw + r * DIN;
        float acc = 0.0f;
#pragma unroll 8
        for (int d = 0; d < DIN; ++d) acc += wr[d] * row[d];
        xdbl[(size_t)bt * RDIM + r] = acc;
    }
}

// ---------------------------------------------------------------------------
// delta[bt][d] = softplus( dt_proj_w[d] . x_dbl[bt][0:24] + dt_proj_b[d] )
// ---------------------------------------------------------------------------
__global__ __launch_bounds__(192) void delta_kernel(
    const float* __restrict__ xdbl, const float* __restrict__ dtw,
    const float* __restrict__ dtb, float* __restrict__ delta)
{
    __shared__ float dtr[DTRANK];
    const int bt = blockIdx.x;
    if (threadIdx.x < DTRANK) dtr[threadIdx.x] = xdbl[(size_t)bt * RDIM + threadIdx.x];
    __syncthreads();
    const int d = threadIdx.x;
    const float* w = dtw + d * DTRANK;
    float acc = dtb[d];
#pragma unroll
    for (int r = 0; r < DTRANK; ++r) acc += w[r] * dtr[r];
    const float sp = (acc > 20.0f) ? acc : log1pf(__expf(acc));
    delta[(size_t)bt * DIN + d] = sp;
}

// ---------------------------------------------------------------------------
// Scan pass 1: local chunk scan from h=0; emit (prod dA, h_final) per state.
// Grid (NCHUNK, BFN, 2 dirs), block 192 (one thread per channel d).
// ---------------------------------------------------------------------------
__global__ __launch_bounds__(192) void scan_pass1(
    const float* __restrict__ delta, const float* __restrict__ xc,
    const float* __restrict__ xdbl,  const float* __restrict__ A_log,
    const float* __restrict__ Ab_log,
    float* __restrict__ chH, float* __restrict__ chP)
{
    const int d     = threadIdx.x;
    const int chunk = blockIdx.x;
    const int bf    = blockIdx.y;
    const int dir   = blockIdx.z;
    const float* alog = dir ? Ab_log : A_log;
    const int tstep = dir ? -1 : 1;

    float Aneg[NSTATE], h[NSTATE], P[NSTATE];
#pragma unroll
    for (int n = 0; n < NSTATE; ++n) {
        Aneg[n] = -__expf(alog[d * NSTATE + n]);
        h[n] = 0.0f; P[n] = 1.0f;
    }

    for (int j = 0; j < CLEN; ++j) {
        const int p = chunk * CLEN + j;               // position in scan order
        const int t = dir ? (TLEN - 1 - p) : p;       // flip for backward
        const size_t bt = (size_t)bf * TLEN + t;
        // prefetch next timestep's operands (global_prefetch_b8)
        if (j + 1 < CLEN) {
            const size_t btn = bt + tstep;
            __builtin_prefetch(delta + btn * DIN + d, 0, 1);
            __builtin_prefetch(xdbl + btn * RDIM + DTRANK, 0, 1);
        }
        const float dl = delta[bt * DIN + d];
        const float du = dl * xc[bt * DIN + d];
        const float* Bp = xdbl + bt * RDIM + DTRANK;  // B rows 24..39 (16B aligned)
        float4 b0 = *(const float4*)(Bp);
        float4 b1 = *(const float4*)(Bp + 4);
        float4 b2 = *(const float4*)(Bp + 8);
        float4 b3 = *(const float4*)(Bp + 12);
        const float Bv[NSTATE] = {b0.x,b0.y,b0.z,b0.w, b1.x,b1.y,b1.z,b1.w,
                                  b2.x,b2.y,b2.z,b2.w, b3.x,b3.y,b3.z,b3.w};
#pragma unroll
        for (int n = 0; n < NSTATE; ++n) {
            const float dA = __expf(dl * Aneg[n]);
            h[n] = dA * h[n] + du * Bv[n];
            P[n] *= dA;
        }
    }
    const size_t base = (((size_t)dir * BFN + bf) * NCHUNK + chunk) * (DIN * NSTATE)
                        + (size_t)d * NSTATE;
#pragma unroll
    for (int n = 0; n < NSTATE; ++n) { chH[base + n] = h[n]; chP[base + n] = P[n]; }
}

// ---------------------------------------------------------------------------
// Scan pass 2: exclusive composition of chunk summaries -> per-chunk h_init.
// One thread per (dir, bf, d, n) = 73728 threads, 16-step serial loop.
// ---------------------------------------------------------------------------
__global__ __launch_bounds__(256) void scan_pass2(
    const float* __restrict__ chH, const float* __restrict__ chP,
    float* __restrict__ chI)
{
    const int tid = blockIdx.x * 256 + threadIdx.x;
    const int total = 2 * BFN * DIN * NSTATE;
    if (tid >= total) return;
    const int dn  = tid % (DIN * NSTATE);
    const int dbf = tid / (DIN * NSTATE);            // dir*BFN + bf
    float h = 0.0f;
    for (int c = 0; c < NCHUNK; ++c) {
        const size_t idx = ((size_t)dbf * NCHUNK + c) * (DIN * NSTATE) + dn;
        chI[idx] = h;
        h = chP[idx] * h + chH[idx];
    }
}

// ---------------------------------------------------------------------------
// Scan pass 3: replay with corrected h_init; emit y = sum_n h_n * C_n.
// yout layout: [dir][bf][t][d]
// ---------------------------------------------------------------------------
__global__ __launch_bounds__(192) void scan_pass3(
    const float* __restrict__ delta, const float* __restrict__ xc,
    const float* __restrict__ xdbl,  const float* __restrict__ A_log,
    const float* __restrict__ Ab_log, const float* __restrict__ chI,
    float* __restrict__ yout)
{
    const int d     = threadIdx.x;
    const int chunk = blockIdx.x;
    const int bf    = blockIdx.y;
    const int dir   = blockIdx.z;
    const float* alog = dir ? Ab_log : A_log;
    const int tstep = dir ? -1 : 1;

    float Aneg[NSTATE], h[NSTATE];
    const size_t ibase = (((size_t)dir * BFN + bf) * NCHUNK + chunk) * (DIN * NSTATE)
                         + (size_t)d * NSTATE;
#pragma unroll
    for (int n = 0; n < NSTATE; ++n) {
        Aneg[n] = -__expf(alog[d * NSTATE + n]);
        h[n] = chI[ibase + n];
    }
    float* ydst = yout + ((size_t)dir * BFN + bf) * (size_t)TLEN * DIN;

    for (int j = 0; j < CLEN; ++j) {
        const int p = chunk * CLEN + j;
        const int t = dir ? (TLEN - 1 - p) : p;
        const size_t bt = (size_t)bf * TLEN + t;
        if (j + 1 < CLEN) {
            const size_t btn = bt + tstep;
            __builtin_prefetch(delta + btn * DIN + d, 0, 1);
            __builtin_prefetch(xdbl + btn * RDIM + DTRANK, 0, 1);
        }
        const float dl = delta[bt * DIN + d];
        const float du = dl * xc[bt * DIN + d];
        const float* Bp = xdbl + bt * RDIM + DTRANK;
        float4 b0 = *(const float4*)(Bp);
        float4 b1 = *(const float4*)(Bp + 4);
        float4 b2 = *(const float4*)(Bp + 8);
        float4 b3 = *(const float4*)(Bp + 12);
        const float* Cp = Bp + NSTATE;                // C rows 40..55
        float4 c0 = *(const float4*)(Cp);
        float4 c1 = *(const float4*)(Cp + 4);
        float4 c2 = *(const float4*)(Cp + 8);
        float4 c3 = *(const float4*)(Cp + 12);
        const float Bv[NSTATE] = {b0.x,b0.y,b0.z,b0.w, b1.x,b1.y,b1.z,b1.w,
                                  b2.x,b2.y,b2.z,b2.w, b3.x,b3.y,b3.z,b3.w};
        const float Cv[NSTATE] = {c0.x,c0.y,c0.z,c0.w, c1.x,c1.y,c1.z,c1.w,
                                  c2.x,c2.y,c2.z,c2.w, c3.x,c3.y,c3.z,c3.w};
        float y = 0.0f;
#pragma unroll
        for (int n = 0; n < NSTATE; ++n) {
            const float dA = __expf(dl * Aneg[n]);
            h[n] = dA * h[n] + du * Bv[n];
            y += h[n] * Cv[n];
        }
        ydst[(size_t)t * DIN + d] = y;
    }
}

// ---------------------------------------------------------------------------
// y_cat[bt][c] = (c<192) ? yfwd + ybwd + 2*u*D[c]  :  z (= xc) channel c-192
// Written directly in bf16 as the out_proj GEMM A operand.
// ---------------------------------------------------------------------------
__global__ __launch_bounds__(256) void assemble_kernel(
    const float* __restrict__ ybuf, const float* __restrict__ xc,
    const float* __restrict__ Dvec, __bf16* __restrict__ ycat)
{
    const size_t idx = (size_t)blockIdx.x * 256 + threadIdx.x;
    const size_t total = (size_t)NROW * GK;
    if (idx >= total) return;
    const int    c  = (int)(idx % GK);
    const size_t bt = idx / GK;
    float v;
    if (c < DIN) {
        const size_t o = bt * DIN + c;
        const float u = xc[o];
        v = ybuf[o] + ybuf[(size_t)BFN * TLEN * DIN + o] + 2.0f * u * Dvec[c];
    } else {
        v = xc[bt * DIN + (c - DIN)];
    }
    ycat[idx] = (__bf16)v;
}

// ---------------------------------------------------------------------------
extern "C" void kernel_launch(void* const* d_in, const int* in_sizes, int n_in,
                              void* d_out, int out_size, void* d_ws, size_t ws_size,
                              hipStream_t stream)
{
    const float* x         = (const float*)d_in[0];
    const float* in_proj_w = (const float*)d_in[1];
    const float* in_proj_b = (const float*)d_in[2];
    const float* conv_w    = (const float*)d_in[3];
    const float* conv_b    = (const float*)d_in[4];
    const float* A_log     = (const float*)d_in[5];
    const float* Ab_log    = (const float*)d_in[6];
    const float* Dvec      = (const float*)d_in[7];
    const float* x_proj_w  = (const float*)d_in[8];
    const float* dt_proj_w = (const float*)d_in[9];
    const float* dt_proj_b = (const float*)d_in[10];
    const float* out_proj_w= (const float*)d_in[11];
    const float* out_proj_b= (const float*)d_in[12];
    float* out = (float*)d_out;

    // Workspace carving (fp32 words unless noted).  Overlays:
    //  - x_bf   (bf16 A of GEMM1) lives in ybuf   (ybuf first written in pass 3)
    //  - ycat_bf(bf16 A of GEMM2) lives in xp     (xp dead after the conv)
    float* ws    = (float*)d_ws;
    float* xp    = ws;                               // NROW*384   = 9,437,184
    float* xc    = xp   + (size_t)NROW * GK;         // NROW*192   = 4,718,592
    float* xdbl  = xc   + (size_t)NROW * DIN;        // NROW*56    = 1,376,256
    float* delta = xdbl + (size_t)NROW * RDIM;       // NROW*192   = 4,718,592
    float* ybuf  = delta+ (size_t)NROW * DIN;        // 2*NROW*192 = 9,437,184
    float* chH   = ybuf + 2 * (size_t)NROW * DIN;    // 1,179,648
    float* chP   = chH  + (size_t)2 * BFN * NCHUNK * DIN * NSTATE;
    float* chI   = chP  + (size_t)2 * BFN * NCHUNK * DIN * NSTATE;
    float* wtail = chI  + (size_t)2 * BFN * NCHUNK * DIN * NSTATE;
    __bf16* w_in_bf  = (__bf16*)wtail;               // 147,456 bf16
    __bf16* w_out_bf = w_in_bf + (size_t)GK * GK;    // 147,456 bf16
    __bf16* x_bf     = (__bf16*)ybuf;                // NROW*384 bf16 (overlay)
    __bf16* ycat_bf  = (__bf16*)xp;                  // NROW*384 bf16 (overlay)
    (void)in_sizes; (void)n_in; (void)out_size; (void)ws_size;

    const size_t NX = (size_t)NROW * GK;             // 9,437,184
    const size_t NW = (size_t)GK * GK;               // 147,456

    // 0) one-time bf16 conversions (x and both weight matrices)
    cvt_bf16_kernel<<<(int)((NX / 8 + 255) / 256), 256, 0, stream>>>(x, x_bf, NX);
    cvt_bf16_kernel<<<(int)((NW / 8 + 255) / 256), 256, 0, stream>>>(in_proj_w,  w_in_bf,  NW);
    cvt_bf16_kernel<<<(int)((NW / 8 + 255) / 256), 256, 0, stream>>>(out_proj_w, w_out_bf, NW);
    // 1) in_proj GEMM (WMMA bf16)
    wmma_gemm384<<<dim3(NROW / 128, GK / 96), 256, 0, stream>>>(x_bf, w_in_bf, in_proj_b, xp);
    // 2) grouped conv3 + SiLU
    conv_silu_kernel<<<NROW, DIN, 0, stream>>>(xp, conv_w, conv_b, xc);
    // 3) x_proj  (dt | B | C)
    xdbl_kernel<<<NROW, 64, 0, stream>>>(xc, x_proj_w, xdbl);
    // 4) dt_proj + softplus
    delta_kernel<<<NROW, DIN, 0, stream>>>(xdbl, dt_proj_w, dt_proj_b, delta);
    // 5-7) chunked bidirectional selective scan (pass 3 overwrites x_bf region)
    scan_pass1<<<dim3(NCHUNK, BFN, 2), DIN, 0, stream>>>(delta, xc, xdbl, A_log, Ab_log, chH, chP);
    scan_pass2<<<(2 * BFN * DIN * NSTATE + 255) / 256, 256, 0, stream>>>(chH, chP, chI);
    scan_pass3<<<dim3(NCHUNK, BFN, 2), DIN, 0, stream>>>(delta, xc, xdbl, A_log, Ab_log, chI, ybuf);
    // 8) assemble [fwd+bwd+2uD | z] -> bf16 (overwrites xp region)
    assemble_kernel<<<(int)(((size_t)NROW * GK + 255) / 256), 256, 0, stream>>>(ybuf, xc, Dvec, ycat_bf);
    // 9) out_proj GEMM (WMMA bf16) -> d_out (fp32)
    wmma_gemm384<<<dim3(NROW / 128, GK / 96), 256, 0, stream>>>(ycat_bf, w_out_bf, out_proj_b, out);
}
